// Net_82222853915377
// MI455X (gfx1250) — matrix operand order repaired and checked by reading, not compiled, not run on hardware
//
#include <hip/hip_runtime.h>

typedef __attribute__((ext_vector_type(16))) _Float16 v16h;
typedef __attribute__((ext_vector_type(8)))  float    v8f;

#define NND 30000
#define NED 180000
#define NGR 64
#define TWO_PI_7 0.8975979010256552f

static inline int cdiv_h(int a, int b) { return (a + b - 1) / b; }

__device__ __forceinline__ v8f wmma16(v16h a, v16h b, v8f c) {
  return __builtin_amdgcn_wmma_f32_16x16x32_f16(false, a, false, b, (short)0, c, false, false);
}

// ---------------- weight prep: conv W (10, FO, FI) -> f16 (FO, KDpad) with Wt[o*KDpad + k*FI+f] ----
__global__ void prep_convw_kernel(const float* __restrict__ W, _Float16* __restrict__ Wt,
                                  int FO, int FI, int KD, int KDpad) {
  int idx = blockIdx.x * blockDim.x + threadIdx.x;
  int total = FO * KDpad;
  if (idx >= total) return;
  int o = idx / KDpad, kk = idx - o * KDpad;
  float v = 0.f;
  if (kk < KD) {
    int k = kk / FI, f = kk - k * FI;
    v = W[(k * FO + o) * FI + f];
  }
  Wt[idx] = (_Float16)v;
}

// lin W (O, K) row-major -> f16 (Opad, Kpad) with Wt[o*Kpad + k] = W[o,k], zero padded
__global__ void prep_lin_kernel(const float* __restrict__ W, _Float16* __restrict__ Wt,
                                int O, int Opad, int K, int Kpad) {
  int idx = blockIdx.x * blockDim.x + threadIdx.x;
  int total = Opad * Kpad;
  if (idx >= total) return;
  int o = idx / Kpad, k = idx - o * Kpad;
  float v = (o < O && k < K) ? W[o * K + k] : 0.f;
  Wt[idx] = (_Float16)v;
}

// ---------------- edge precompute: Pf (E,10) and transport trig (E,4) ----------------
__global__ void edge_precomp_kernel(const float* __restrict__ coords, const float* __restrict__ conn,
                                    float* __restrict__ Pf, float* __restrict__ trig, int E) {
  int e = blockIdx.x * blockDim.x + threadIdx.x;
  if (e >= E) return;
  float r  = coords[e * 2 + 0];
  float th = coords[e * 2 + 1];
  float t = fminf(fmaxf(r, 0.f), 1.f);
  float s1, c1, s2, c2;
  sincosf(th, &s1, &c1);
  sincosf(2.f * th, &s2, &c2);
  float ang[5] = {1.f, c1, s1, c2, s2};
  float rad[2] = {1.f - t, t};
#pragma unroll
  for (int k = 0; k < 10; ++k) Pf[e * 10 + k] = rad[k / 5] * ang[k % 5];
  float thc = conn[e];
  float cs1, cc1, cs2, cc2;
  sincosf(thc, &cs1, &cc1);
  sincosf(2.f * thc, &cs2, &cc2);
  trig[e * 4 + 0] = cc1;
  trig[e * 4 + 1] = cs1;
  trig[e * 4 + 2] = cc2;
  trig[e * 4 + 3] = cs2;
}

// ---------------- gem_conv via WMMA ----------------
// One wave handles a 16-edge tile.  A[m, kk] = P[m, kk/F_IN] * xf[m, kk%F_IN] built per-lane
// in contiguous k order (k = 16*(lane/16) + s); B tiles loaded as contiguous v16h from (F_OUT x KDPAD) f16
// weights; 16x16 f32 accumulators scattered with global atomics (segment_sum over dst).
template <int C_IN, int D_IN, int F_OUT, int IN_ORDER>
__global__ void __launch_bounds__(128)
gem_conv_kernel(const float* __restrict__ X, const _Float16* __restrict__ Wt,
                const int* __restrict__ dstIdx, const int* __restrict__ srcIdx,
                const float* __restrict__ Pf, const float* __restrict__ trig,
                float* __restrict__ out, int nEdges) {
  constexpr int F_IN = C_IN * D_IN;
  constexpr int KD = 10 * F_IN;
  constexpr int KDPAD = (KD + 31) & ~31;
  constexpr int NT = F_OUT / 16;
  constexpr int WV = 4;

  __shared__ float Xl[WV][16][F_IN];
  __shared__ float Pl[WV][16][10];
  __shared__ float Tl[WV][16][4];
  __shared__ int   Sl[WV][16];
  __shared__ int   Dl[WV][16];

  const int lane = threadIdx.x & 31;
  const int wid  = threadIdx.x >> 5;
  const int tile = blockIdx.x * WV + wid;

  // phase A: per-edge metadata
  if (lane < 16) {
    int e = tile * 16 + lane;
    bool v = (e < nEdges);
    Sl[wid][lane] = v ? srcIdx[e] : 0;
    Dl[wid][lane] = v ? dstIdx[e] : 0;
#pragma unroll
    for (int i = 0; i < 10; ++i) Pl[wid][lane][i] = v ? Pf[e * 10 + i] : 0.f;
#pragma unroll
    for (int i = 0; i < 4; ++i)  Tl[wid][lane][i] = v ? trig[e * 4 + i] : 0.f;
  }
  __syncthreads();

  // phase B: gather x[src] + transport rotation into LDS
  for (int idx = lane; idx < 16 * F_IN; idx += 32) {
    int m = idx / F_IN, f = idx - m * F_IN;
    int e = tile * 16 + m;
    float val = 0.f;
    if (e < nEdges) {
      const float* xp = X + (size_t)Sl[wid][m] * F_IN;
      if (IN_ORDER == 0) {
        val = xp[f];
      } else {
        int d = f % D_IN;
        float c1 = Tl[wid][m][0], s1 = Tl[wid][m][1];
        float c2 = Tl[wid][m][2], s2 = Tl[wid][m][3];
        if      (d == 0) val = xp[f];
        else if (d == 1) val = xp[f] * c1 - xp[f + 1] * s1;
        else if (d == 2) val = xp[f - 1] * s1 + xp[f] * c1;
        else if (d == 3) val = xp[f] * c2 - xp[f + 1] * s2;
        else             val = xp[f - 1] * s2 + xp[f] * c2;
      }
    }
    Xl[wid][m][f] = val;
  }
  __syncthreads();

  const int g = lane >> 4, l15 = lane & 15;
  v8f acc[NT] = {};

#pragma unroll 1
  for (int kc = 0; kc < KD; kc += 32) {
    // A fragment: K = 16*g + 2*j + h
    v16h a;
#pragma unroll
    for (int j = 0; j < 8; ++j)
#pragma unroll
      for (int h = 0; h < 2; ++h) {
        int kk = kc + (g << 4) + (j << 1) + h;
        float av = 0.f;
        if (kk < KD) {
          int kr = kk / F_IN, fr = kk - kr * F_IN;
          av = Pl[wid][l15][kr] * Xl[wid][l15][fr];
        }
        a[(j << 1) + h] = (_Float16)av;
      }
#pragma unroll
    for (int nt = 0; nt < NT; ++nt) {
      // B fragment: K = 16*g + 2*j + h -> 16 contiguous halves at row (nt*16+l15)
      v16h b = *(const v16h*)(Wt + (size_t)(nt * 16 + l15) * KDPAD + kc + (g << 4));
      acc[nt] = wmma16(a, b, acc[nt]);
    }
  }

  // scatter-add msg rows to out[dst]
#pragma unroll
  for (int nt = 0; nt < NT; ++nt)
#pragma unroll
    for (int j = 0; j < 8; ++j) {
      int row = (g << 3) + j;
      int e = tile * 16 + row;
      if (e < nEdges)
        atomicAdd(&out[(size_t)Dl[wid][row] * F_OUT + nt * 16 + l15], acc[nt][j]);
    }
}

// ---------------- block1 shortcut: sc[n,o,d] = sum_c W[ord(d),o,c] * x[n,c,d] ----------------
__global__ void sc1_kernel(const float* __restrict__ x, const float* __restrict__ Wsc,
                           float* __restrict__ sc, int N) {
  int idx = blockIdx.x * blockDim.x + threadIdx.x;
  if (idx >= N * 80) return;
  int n = idx / 80, rem = idx - n * 80, o = rem / 5, d = rem - o * 5;
  const int ord[5] = {0, 1, 1, 2, 2};
  const float* w = Wsc + (ord[d] * 16 + o) * 7;
  const float* xp = x + n * 35 + d;
  float s = 0.f;
#pragma unroll
  for (int c = 0; c < 7; ++c) s += w[c] * xp[c * 5];
  sc[idx] = s;
}

// ---------------- block3 shortcut: sc16[n,o] = sum_c b3sc_W[o,c] * x[n,c,0] ----------------
__global__ void sc3_kernel(const float* __restrict__ xin, const float* __restrict__ W,
                           float* __restrict__ sc, int N) {
  int idx = blockIdx.x * blockDim.x + threadIdx.x;
  if (idx >= N * 16) return;
  int n = idx / 16, o = idx - n * 16;
  float s = 0.f;
#pragma unroll
  for (int c = 0; c < 16; ++c) s += W[o * 16 + c] * xin[n * 80 + c * 5];
  sc[idx] = s;
}

// ---------------- post (order 2): bias on comp0, +residual, Fourier reg_relu ----------------
__global__ void post2_kernel(const float* __restrict__ h, const float* __restrict__ bias,
                             const float* __restrict__ res, float* __restrict__ out, int N) {
  int idx = blockIdx.x * blockDim.x + threadIdx.x;
  if (idx >= N * 16) return;
  int n = idx / 16, c = idx - n * 16;
  const float* hp = h + n * 80 + c * 5;
  float v[5];
#pragma unroll
  for (int d = 0; d < 5; ++d) v[d] = hp[d];
  v[0] += bias[c];
  if (res) {
    const float* rp = res + n * 80 + c * 5;
#pragma unroll
    for (int d = 0; d < 5; ++d) v[d] += rp[d];
  }
  float o5[5] = {0.f, 0.f, 0.f, 0.f, 0.f};
#pragma unroll
  for (int i = 0; i < 7; ++i) {
    float phi = (float)i * TWO_PI_7;
    float s1, c1, s2, c2;
    sincosf(phi, &s1, &c1);
    sincosf(2.f * phi, &s2, &c2);
    float s = v[0] + v[1] * c1 + v[2] * s1 + v[3] * c2 + v[4] * s2;
    s = fmaxf(s, 0.f);
    const float inv = 1.f / 7.f;
    o5[0] += inv * s;
    o5[1] += 2.f * inv * c1 * s;
    o5[2] += 2.f * inv * s1 * s;
    o5[3] += 2.f * inv * c2 * s;
    o5[4] += 2.f * inv * s2 * s;
  }
  float* op = out + n * 80 + c * 5;
#pragma unroll
  for (int d = 0; d < 5; ++d) op[d] = o5[d];
}

// ---------------- post (order 0): bias, +residual, relu ----------------
__global__ void post0_kernel(const float* __restrict__ h, const float* __restrict__ bias,
                             const float* __restrict__ res, float* __restrict__ out, int N) {
  int idx = blockIdx.x * blockDim.x + threadIdx.x;
  if (idx >= N * 16) return;
  int c = idx & 15;
  float v = h[idx] + bias[c];
  if (res) v += res[idx];
  out[idx] = fmaxf(v, 0.f);
}

// ---------------- graph node counts ----------------
__global__ void counts_kernel(const int* __restrict__ batch, float* __restrict__ counts, int N) {
  int n = blockIdx.x * blockDim.x + threadIdx.x;
  if (n < N) atomicAdd(&counts[batch[n]], 1.0f);
}

// ---------------- MLP head: (16->256 relu ->40) per 16-node tile, WMMA both stages ----------------
__global__ void __launch_bounds__(128)
head_kernel(const float* __restrict__ xin, const _Float16* __restrict__ W1,
            const float* __restrict__ b1, const _Float16* __restrict__ W2,
            const float* __restrict__ b2, const int* __restrict__ batch,
            float* __restrict__ pooled, int N) {
  constexpr int WV = 4;
  __shared__ _Float16 vT[WV][16][16];
  __shared__ _Float16 fT[WV][16][256];
  __shared__ int bT[WV][16];

  const int lane = threadIdx.x & 31;
  const int wid  = threadIdx.x >> 5;
  const int tile = blockIdx.x * WV + wid;

  for (int idx = lane; idx < 256; idx += 32) {
    int m = idx >> 4, c = idx & 15;
    int node = tile * 16 + m;
    vT[wid][m][c] = (_Float16)((node < N) ? xin[node * 16 + c] : 0.f);
  }
  if (lane < 16) {
    int node = tile * 16 + lane;
    bT[wid][lane] = (node < N) ? batch[node] : 0;
  }
  __syncthreads();

  const int g = lane >> 4, l15 = lane & 15;

  // stage 1: f = relu(x @ W1^T + b1), K=16 (padded 32)
  v16h a;
#pragma unroll
  for (int j = 0; j < 8; ++j)
#pragma unroll
    for (int h = 0; h < 2; ++h) {
      int kk = (g << 4) + (j << 1) + h;
      a[(j << 1) + h] = (kk < 16) ? vT[wid][l15][kk] : (_Float16)0.f;
    }
#pragma unroll 1
  for (int nt = 0; nt < 16; ++nt) {
    v16h b = *(const v16h*)(W1 + (size_t)(nt * 16 + l15) * 32 + (g << 4));
    v8f c = {};
    c = wmma16(a, b, c);
#pragma unroll
    for (int j = 0; j < 8; ++j) {
      int row = (g << 3) + j, col = nt * 16 + l15;
      fT[wid][row][col] = (_Float16)fmaxf(c[j] + b1[col], 0.f);
    }
  }
  __syncthreads();

  // stage 2: logits = f @ W2^T + b2, K=256, 40 cols (padded 48)
  v8f acc[3] = {};
#pragma unroll 1
  for (int kc = 0; kc < 256; kc += 32) {
    v16h a2;
#pragma unroll
    for (int j = 0; j < 8; ++j)
#pragma unroll
      for (int h = 0; h < 2; ++h) {
        int kk = kc + (g << 4) + (j << 1) + h;
        a2[(j << 1) + h] = fT[wid][l15][kk];
      }
#pragma unroll
    for (int nt = 0; nt < 3; ++nt) {
      v16h b2v = *(const v16h*)(W2 + (size_t)(nt * 16 + l15) * 256 + kc + (g << 4));
      acc[nt] = wmma16(a2, b2v, acc[nt]);
    }
  }
#pragma unroll
  for (int nt = 0; nt < 3; ++nt)
#pragma unroll
    for (int j = 0; j < 8; ++j) {
      int row = (g << 3) + j;
      int node = tile * 16 + row;
      int col = nt * 16 + l15;
      if (node < N && col < 40)
        atomicAdd(&pooled[bT[wid][row] * 40 + col], acc[nt][j] + b2[col]);
    }
}

// ---------------- final: mean pool + log_softmax ----------------
__global__ void final_kernel(const float* __restrict__ pooled, const float* __restrict__ counts,
                             float* __restrict__ outp) {
  int gph = threadIdx.x;
  if (gph >= NGR) return;
  float cnt = fmaxf(counts[gph], 1.f);
  float v[40];
  float mx = -1e30f;
#pragma unroll
  for (int o = 0; o < 40; ++o) {
    v[o] = pooled[gph * 40 + o] / cnt;
    mx = fmaxf(mx, v[o]);
  }
  float s = 0.f;
#pragma unroll
  for (int o = 0; o < 40; ++o) s += __expf(v[o] - mx);
  float lse = mx + __logf(s);
#pragma unroll
  for (int o = 0; o < 40; ++o) outp[gph * 40 + o] = v[o] - lse;
}

extern "C" void kernel_launch(void* const* d_in, const int* in_sizes, int n_in,
                              void* d_out, int out_size, void* d_ws, size_t ws_size,
                              hipStream_t stream) {
  const float* x        = (const float*)d_in[0];
  const float* coords   = (const float*)d_in[1];
  const float* conn     = (const float*)d_in[2];
  const float* b1c1_W   = (const float*)d_in[3];
  const float* b1c1_b   = (const float*)d_in[4];
  const float* b1c2_W   = (const float*)d_in[5];
  const float* b1c2_b   = (const float*)d_in[6];
  const float* b1sc_W   = (const float*)d_in[7];
  const float* b2c1_W   = (const float*)d_in[8];
  const float* b2c1_b   = (const float*)d_in[9];
  const float* b2c2_W   = (const float*)d_in[10];
  const float* b2c2_b   = (const float*)d_in[11];
  const float* b3c1_W   = (const float*)d_in[12];
  const float* b3c1_b   = (const float*)d_in[13];
  const float* b3c2_W   = (const float*)d_in[14];
  const float* b3c2_b   = (const float*)d_in[15];
  const float* b3sc_W   = (const float*)d_in[16];
  const float* lin1_W   = (const float*)d_in[17];
  const float* lin1_b   = (const float*)d_in[18];
  const float* lin2_W   = (const float*)d_in[19];
  const float* lin2_b   = (const float*)d_in[20];
  const int*   ei       = (const int*)d_in[21];
  const int*   batch    = (const int*)d_in[22];
  float* outp = (float*)d_out;

  const int N = NND, E = NED;
  const int* dstI = ei;
  const int* srcI = ei + E;

  // workspace carve-up
  char* wsp = (char*)d_ws;
  auto alloc = [&](size_t bytes) -> char* {
    char* p = wsp;
    wsp += (bytes + 255) & ~(size_t)255;
    return p;
  };
  float* Pf    = (float*)alloc((size_t)E * 10 * 4);
  float* trig  = (float*)alloc((size_t)E * 4 * 4);
  float* bufA  = (float*)alloc((size_t)N * 80 * 4);
  float* bufB  = (float*)alloc((size_t)N * 80 * 4);
  float* bufC  = (float*)alloc((size_t)N * 80 * 4);
  float* pooled = (float*)alloc((NGR * 40 + NGR) * 4);
  float* counts = pooled + NGR * 40;
  _Float16* wt_b1c1 = (_Float16*)alloc((size_t)80 * 352 * 2);
  _Float16* wt_b1c2 = (_Float16*)alloc((size_t)80 * 800 * 2);
  _Float16* wt_b2c1 = (_Float16*)alloc((size_t)80 * 800 * 2);
  _Float16* wt_b2c2 = (_Float16*)alloc((size_t)80 * 800 * 2);
  _Float16* wt_b3c1 = (_Float16*)alloc((size_t)16 * 800 * 2);
  _Float16* wt_b3c2 = (_Float16*)alloc((size_t)16 * 160 * 2);
  _Float16* wt_l1   = (_Float16*)alloc((size_t)256 * 32 * 2);
  _Float16* wt_l2   = (_Float16*)alloc((size_t)48 * 256 * 2);

  // ---- weight prep ----
  prep_convw_kernel<<<cdiv_h(80 * 352, 256), 256, 0, stream>>>(b1c1_W, wt_b1c1, 80, 35, 350, 352);
  prep_convw_kernel<<<cdiv_h(80 * 800, 256), 256, 0, stream>>>(b1c2_W, wt_b1c2, 80, 80, 800, 800);
  prep_convw_kernel<<<cdiv_h(80 * 800, 256), 256, 0, stream>>>(b2c1_W, wt_b2c1, 80, 80, 800, 800);
  prep_convw_kernel<<<cdiv_h(80 * 800, 256), 256, 0, stream>>>(b2c2_W, wt_b2c2, 80, 80, 800, 800);
  prep_convw_kernel<<<cdiv_h(16 * 800, 256), 256, 0, stream>>>(b3c1_W, wt_b3c1, 16, 80, 800, 800);
  prep_convw_kernel<<<cdiv_h(16 * 160, 256), 256, 0, stream>>>(b3c2_W, wt_b3c2, 16, 16, 160, 160);
  prep_lin_kernel<<<cdiv_h(256 * 32, 256), 256, 0, stream>>>(lin1_W, wt_l1, 256, 256, 16, 32);
  prep_lin_kernel<<<cdiv_h(48 * 256, 256), 256, 0, stream>>>(lin2_W, wt_l2, 40, 48, 256, 256);

  // ---- edge precompute ----
  edge_precomp_kernel<<<cdiv_h(E, 256), 256, 0, stream>>>(coords, conn, Pf, trig, E);

  const int edgeTiles = cdiv_h(E, 16);
  const int convBlocks = cdiv_h(edgeTiles, 4);
  const int nc16 = cdiv_h(N * 16, 256);

  // ---- block 1 ----
  sc1_kernel<<<cdiv_h(N * 80, 256), 256, 0, stream>>>(x, b1sc_W, bufC, N);
  hipMemsetAsync(bufA, 0, (size_t)N * 80 * 4, stream);
  gem_conv_kernel<7, 5, 80, 2><<<convBlocks, 128, 0, stream>>>(x, wt_b1c1, dstI, srcI, Pf, trig, bufA, E);
  post2_kernel<<<nc16, 256, 0, stream>>>(bufA, b1c1_b, nullptr, bufB, N);
  hipMemsetAsync(bufA, 0, (size_t)N * 80 * 4, stream);
  gem_conv_kernel<16, 5, 80, 2><<<convBlocks, 128, 0, stream>>>(bufB, wt_b1c2, dstI, srcI, Pf, trig, bufA, E);
  post2_kernel<<<nc16, 256, 0, stream>>>(bufA, b1c2_b, bufC, bufB, N);

  // ---- block 2 ----
  hipMemsetAsync(bufA, 0, (size_t)N * 80 * 4, stream);
  gem_conv_kernel<16, 5, 80, 2><<<convBlocks, 128, 0, stream>>>(bufB, wt_b2c1, dstI, srcI, Pf, trig, bufA, E);
  post2_kernel<<<nc16, 256, 0, stream>>>(bufA, b2c1_b, nullptr, bufC, N);
  hipMemsetAsync(bufA, 0, (size_t)N * 80 * 4, stream);
  gem_conv_kernel<16, 5, 80, 2><<<convBlocks, 128, 0, stream>>>(bufC, wt_b2c2, dstI, srcI, Pf, trig, bufA, E);
  post2_kernel<<<nc16, 256, 0, stream>>>(bufA, b2c2_b, bufB, bufB, N);

  // ---- block 3 ----
  sc3_kernel<<<nc16, 256, 0, stream>>>(bufB, b3sc_W, bufC, N);
  hipMemsetAsync(bufA, 0, (size_t)N * 16 * 4, stream);
  gem_conv_kernel<16, 5, 16, 2><<<convBlocks, 128, 0, stream>>>(bufB, wt_b3c1, dstI, srcI, Pf, trig, bufA, E);
  post0_kernel<<<nc16, 256, 0, stream>>>(bufA, b3c1_b, nullptr, bufB, N);
  hipMemsetAsync(bufA, 0, (size_t)N * 16 * 4, stream);
  gem_conv_kernel<16, 1, 16, 0><<<convBlocks, 128, 0, stream>>>(bufB, wt_b3c2, dstI, srcI, Pf, trig, bufA, E);
  post0_kernel<<<nc16, 256, 0, stream>>>(bufA, b3c2_b, bufC, bufB, N);

  // ---- head + pooling ----
  hipMemsetAsync(pooled, 0, (NGR * 40 + NGR) * 4, stream);
  counts_kernel<<<cdiv_h(N, 256), 256, 0, stream>>>(batch, counts, N);
  const int nodeTiles = cdiv_h(N, 16);
  head_kernel<<<cdiv_h(nodeTiles, 4), 128, 0, stream>>>(bufB, wt_l1, lin1_b, wt_l2, lin2_b, batch, pooled, N);
  final_kernel<<<1, 64, 0, stream>>>(pooled, counts, outp);
}
